// Encoder_21019569947157
// MI455X (gfx1250) — matrix operand order repaired and checked by reading, not compiled, not run on hardware
//
#include <hip/hip_runtime.h>
#include <hip/hip_bf16.h>

// ---------------------------------------------------------------------------
// Bidirectional tanh-RNN encoder for MI455X (gfx1250, wave32, WMMA).
//
//   Phase 0 (stream):  pre-pack x / W_ih into bf16 hi/lo pairs (one pass),
//                      x kept in linear k-order (contiguous A-fragment order)
//                      so GEMM fragments are single contiguous v16bf loads.
//   Phase 1 (matrix):  u[dir] = x(@rev) @ W_ih^T + (b_ih + b_hh)
//                      -> v_wmma_f32_16x16x32_bf16, hi/lo split (3 WMMA/K=32)
//                      into 3 independent accumulators (no WMMA RAW chain).
//   Phase 2 (latency): h_t = tanh(W_hh h_{t-1} + u_t), 8192 dependent steps
//                      -> persistent kernel, W_hh sharded 32 rows/WG in LDS,
//                      one release/acquire barrier per step, double-buffered h,
//                      both directions concurrent via gridDim.y = 2.
// ---------------------------------------------------------------------------

#define SEQ  8192
#define DIN  1024
#define DH   1024

typedef __bf16 bf16;
typedef bf16  v16bf __attribute__((ext_vector_type(16)));
typedef bf16  v4bf  __attribute__((ext_vector_type(4)));
typedef float v8f   __attribute__((ext_vector_type(8)));

__device__ __forceinline__ v8f wmma_bf16(v16bf a, v16bf b, v8f c) {
    return __builtin_amdgcn_wmma_f32_16x16x32_bf16(false, a, false, b,
                                                   (short)0, c, false, false);
}

// ====================== Phase 0: hi/lo bf16 packing ========================
// Linear layout (for W_ih: row n contiguous in k, matches B-fragment order).
__global__ __launch_bounds__(256) void pack_hilo_lin(
    const float* __restrict__ src, bf16* __restrict__ hi,
    bf16* __restrict__ lo, int n4)
{
    int i = blockIdx.x * 256 + threadIdx.x;
    if (i >= n4) return;
    float4 v = ((const float4*)src)[i];
    bf16 h0 = (bf16)v.x, h1 = (bf16)v.y, h2 = (bf16)v.z, h3 = (bf16)v.w;
    v4bf hv = {h0, h1, h2, h3};
    v4bf lv = {(bf16)(v.x - (float)h0), (bf16)(v.y - (float)h1),
               (bf16)(v.z - (float)h2), (bf16)(v.w - (float)h3)};
    ((v4bf*)hi)[i] = hv;
    ((v4bf*)lo)[i] = lv;
}

// x pack in linear k-order per 32 (contiguous A-fragment order, matching B):
//   k 0..31 -> 0..31
// so lane<16 reads fragment elems [0,16) and lane>=16 reads [16,32) contiguously.
__global__ __launch_bounds__(256) void pack_x_perm(
    const float* __restrict__ x, bf16* __restrict__ hi, bf16* __restrict__ lo)
{
    size_t blk = (size_t)blockIdx.x * 256 + threadIdx.x;  // one 32-elem block
    if (blk >= (size_t)SEQ * DIN / 32) return;
    const float* s = x + blk * 32;
    v16bf h0 = {}, h1 = {}, l0 = {}, l1 = {};
#pragma unroll
    for (int e = 0; e < 32; ++e) {
        int p = e;
        float f = s[e];
        bf16 hb = (bf16)f, lb = (bf16)(f - (float)hb);
        if (p < 16) { h0[p] = hb; l0[p] = lb; }
        else        { h1[p - 16] = hb; l1[p - 16] = lb; }
    }
    *(v16bf*)(hi + blk * 32)      = h0;
    *(v16bf*)(hi + blk * 32 + 16) = h1;
    *(v16bf*)(lo + blk * 32)      = l0;
    *(v16bf*)(lo + blk * 32 + 16) = l1;
}

// ====================== Phase 1: WMMA GEMM (packed) ========================
// One wave = one 16x16 output tile; 8 waves/block share the M-tile (A reuse
// through L1/L2). Inner loop: 4 contiguous v16bf loads + 3 independent WMMAs.
__global__ __launch_bounds__(256) void proj_gemm_packed(
    const bf16* __restrict__ xhi, const bf16* __restrict__ xlo, // [SEQ][DIN] linear
    const bf16* __restrict__ whi, const bf16* __restrict__ wlo, // [DH][DIN] linear
    const float* __restrict__ b1, const float* __restrict__ b2,
    float* __restrict__ u, int reverse)
{
    const int lane   = threadIdx.x & 31;
    const int waveId = blockIdx.x * 8 + (threadIdx.x >> 5);
    const int ntile  = waveId & 63;     // DH/16
    const int mtile  = waveId >> 6;     // SEQ/16

    const int mrow = (mtile << 4) + (lane & 15);
    const int xrow = reverse ? (SEQ - 1 - mrow) : mrow;
    const int ncol = (ntile << 4) + (lane & 15);
    const int fo   = (lane >> 4) << 4;  // fragment half: 0 or 16 elements

    const bf16* xh = xhi + (size_t)xrow * DIN + fo;
    const bf16* xl = xlo + (size_t)xrow * DIN + fo;
    const bf16* wh = whi + (size_t)ncol * DIN + fo;
    const bf16* wl = wlo + (size_t)ncol * DIN + fo;

    v8f acc_hh = {}, acc_lh = {}, acc_hl = {};
    for (int k0 = 0; k0 < DIN; k0 += 32) {
        v16bf ah = *(const v16bf*)(xh + k0);
        v16bf al = *(const v16bf*)(xl + k0);
        v16bf bh = *(const v16bf*)(wh + k0);
        v16bf bl = *(const v16bf*)(wl + k0);
        acc_hh = wmma_bf16(ah, bh, acc_hh);   // independent accumulators:
        acc_lh = wmma_bf16(al, bh, acc_lh);   // no WMMA->WMMA RAW hazard
        acc_hl = wmma_bf16(ah, bl, acc_hl);
    }

    // C/D layout: VGPR v -> M = v + 8*(lane>=16), N = lane&15.
    const float bias  = b1[ncol] + b2[ncol];
    const int   rbase = (mtile << 4) + ((lane >> 4) << 3);
#pragma unroll
    for (int v = 0; v < 8; ++v)
        u[(size_t)(rbase + v) * DH + ncol] =
            (acc_hh[v] + acc_lh[v]) + acc_hl[v] + bias;
}

// ============== Phase 1 fallback: convert-in-loop WMMA GEMM ================
__global__ __launch_bounds__(256) void proj_gemm_cvt(
    const float* __restrict__ x, const float* __restrict__ w,
    const float* __restrict__ b1, const float* __restrict__ b2,
    float* __restrict__ u, int reverse)
{
    const int lane   = threadIdx.x & 31;
    const int waveId = blockIdx.x * 8 + (threadIdx.x >> 5);
    const int ntile  = waveId & 63;
    const int mtile  = waveId >> 6;

    const int mrow = (mtile << 4) + (lane & 15);
    const int xrow = reverse ? (SEQ - 1 - mrow) : mrow;
    const float* xp = x + (size_t)xrow * DIN;
    const int ncol = (ntile << 4) + (lane & 15);
    const float* wp = w + (size_t)ncol * DIN;
    const int kA = (lane >> 4) << 3;
    const int kB = (lane >> 4) << 4;

    v8f acc_hh = {}, acc_lh = {}, acc_hl = {};
    for (int k0 = 0; k0 < DIN; k0 += 32) {
        float a[16], bb[16];
#pragma unroll
        for (int e = 0; e < 8; ++e) {
            a[e]     = xp[k0 + kB + e];
            a[8 + e] = xp[k0 + kB + 8 + e];
        }
#pragma unroll
        for (int e = 0; e < 16; ++e) bb[e] = wp[k0 + kB + e];
        v16bf ahi, alo, bhi, blo;
#pragma unroll
        for (int e = 0; e < 16; ++e) {
            bf16 h = (bf16)a[e];
            ahi[e] = h;  alo[e] = (bf16)(a[e] - (float)h);
            bf16 g = (bf16)bb[e];
            bhi[e] = g;  blo[e] = (bf16)(bb[e] - (float)g);
        }
        acc_hh = wmma_bf16(ahi, bhi, acc_hh);
        acc_lh = wmma_bf16(alo, bhi, acc_lh);
        acc_hl = wmma_bf16(ahi, blo, acc_hl);
    }
    const float bias  = b1[ncol] + b2[ncol];
    const int   rbase = (mtile << 4) + ((lane >> 4) << 3);
#pragma unroll
    for (int v = 0; v < 8; ++v)
        u[(size_t)(rbase + v) * DH + ncol] =
            (acc_hh[v] + acc_lh[v]) + acc_hl[v] + bias;
}

// ========================== Phase 2: recurrence ============================
#define NWG   32          // workgroups per direction
#define ROWS  32          // W_hh rows owned per workgroup
#define WSTR  1040        // padded LDS row stride -> conflict-free W reads
#define SCAN_LDS ((ROWS * WSTR + DH) * sizeof(float))

__global__ __launch_bounds__(256) void rnn_scan(
    const float* __restrict__ fw_hh,
    const float* __restrict__ bw_hh,
    const float* __restrict__ u,      // [2][SEQ][DH]
    float* __restrict__ out,          // [SEQ][2*DH]
    float* __restrict__ hbuf,         // [2][2][DH], zeroed before launch
    unsigned* __restrict__ cnt)       // [2][SEQ], zeroed before launch
{
    extern __shared__ float smem[];
    float* wlds = smem;               // ROWS x WSTR
    float* hsh  = smem + ROWS * WSTR; // DH staged h

    const int dir = blockIdx.y;
    const int wg  = blockIdx.x;
    const int tid = threadIdx.x;

    const float* whh = dir ? bw_hh : fw_hh;
    const float* ud  = u + (size_t)dir * SEQ * DH;
    float*    hb = hbuf + dir * 2 * DH;
    unsigned* cd = cnt  + dir * SEQ;
    const int row0 = wg * ROWS;

    for (int idx = tid; idx < ROWS * DH; idx += 256) {
        int r = idx >> 10, k = idx & (DH - 1);
        wlds[r * WSTR + k] = whh[(size_t)(row0 + r) * DH + k];
    }
    __syncthreads();

    const int row  = tid >> 3;
    const int sub  = tid & 7;
    const int grow = row0 + row;
    const float* wr = wlds + row * WSTR + sub * 4;

    for (int t = 0; t < SEQ; ++t) {
        float uval = ud[(size_t)t * DH + grow];  // independent of h: load early

        if (t) {
            if (tid == 0) {
                while (__hip_atomic_load(&cd[t - 1], __ATOMIC_ACQUIRE,
                                         __HIP_MEMORY_SCOPE_AGENT) < NWG)
                    __builtin_amdgcn_s_sleep(1);
            }
            __syncthreads();
        }

        const float* hin = hb + (t & 1) * DH;
        for (int i = tid; i < DH; i += 256) hsh[i] = hin[i];
        __syncthreads();

        float p = 0.f;
#pragma unroll 8
        for (int j = 0; j < DH / 32; ++j) {
            const float4 wv = *(const float4*)(wr + j * 32);
            const float4 hv = *(const float4*)(hsh + j * 32 + sub * 4);
            p += wv.x * hv.x + wv.y * hv.y + wv.z * hv.z + wv.w * hv.w;
        }
        p += __shfl_xor(p, 1, 8);
        p += __shfl_xor(p, 2, 8);
        p += __shfl_xor(p, 4, 8);

        if (sub == 0) {
            float hn = tanhf(p + uval);
            hb[((t + 1) & 1) * DH + grow] = hn;
            size_t orow = dir ? (size_t)(SEQ - 1 - t) : (size_t)t;
            out[orow * (2 * DH) + (size_t)dir * DH + grow] = hn;
        }

        __threadfence();
        __syncthreads();
        if (tid == 0)
            __hip_atomic_fetch_add(&cd[t], 1u, __ATOMIC_RELEASE,
                                   __HIP_MEMORY_SCOPE_AGENT);
    }
}

// =============================== Launch ====================================
extern "C" void kernel_launch(void* const* d_in, const int* in_sizes, int n_in,
                              void* d_out, int out_size, void* d_ws, size_t ws_size,
                              hipStream_t stream) {
    const float* x     = (const float*)d_in[0];
    const float* fw_ih = (const float*)d_in[1];
    const float* fb_ih = (const float*)d_in[2];
    const float* fw_hh = (const float*)d_in[3];
    const float* fb_hh = (const float*)d_in[4];
    const float* bw_ih = (const float*)d_in[5];
    const float* bb_ih = (const float*)d_in[6];
    const float* bw_hh = (const float*)d_in[7];
    const float* bb_hh = (const float*)d_in[8];
    float* out = (float*)d_out;

    const int gemm_blocks = (SEQ / 16) * (DH / 16) / 8;   // 4096

    // Workspace layout (floats): u_f | u_b | [packed bf16 buffers] | hbuf | cnt
    float* ws  = (float*)d_ws;
    float* u_f = ws;
    float* u_b = ws + (size_t)SEQ * DH;
    float* tail = ws + (size_t)2 * SEQ * DH;           // after u buffers

    const size_t packed_floats =
        (size_t)SEQ * DIN       // xhi+xlo (2 bf16 = 1 float each elem)
      + (size_t)2 * DH * DIN;   // {fw,bw} hi+lo
    const size_t need_packed =
        ((size_t)2 * SEQ * DH + packed_floats) * sizeof(float)
      + 4 * DH * sizeof(float) + 2 * SEQ * sizeof(unsigned);

    float* hbuf;
    unsigned* cnt;

    if (ws_size >= need_packed) {
        bf16* xhi  = (bf16*)tail;
        bf16* xlo  = xhi  + (size_t)SEQ * DIN;
        bf16* fwhi = xlo  + (size_t)SEQ * DIN;
        bf16* fwlo = fwhi + (size_t)DH * DIN;
        bf16* bwhi = fwlo + (size_t)DH * DIN;
        bf16* bwlo = bwhi + (size_t)DH * DIN;
        hbuf = (float*)(bwlo + (size_t)DH * DIN);
        cnt  = (unsigned*)(hbuf + 4 * DH);
        hipMemsetAsync(hbuf, 0,
                       4 * DH * sizeof(float) + 2 * SEQ * sizeof(unsigned),
                       stream);

        pack_x_perm<<<(SEQ * DIN / 32 + 255) / 256, 256, 0, stream>>>(x, xhi, xlo);
        pack_hilo_lin<<<(DH * DIN / 4 + 255) / 256, 256, 0, stream>>>(
            fw_ih, fwhi, fwlo, DH * DIN / 4);
        pack_hilo_lin<<<(DH * DIN / 4 + 255) / 256, 256, 0, stream>>>(
            bw_ih, bwhi, bwlo, DH * DIN / 4);

        proj_gemm_packed<<<gemm_blocks, 256, 0, stream>>>(
            xhi, xlo, fwhi, fwlo, fb_ih, fb_hh, u_f, 0);
        proj_gemm_packed<<<gemm_blocks, 256, 0, stream>>>(
            xhi, xlo, bwhi, bwlo, bb_ih, bb_hh, u_b, 1);
    } else {
        hbuf = tail;
        cnt  = (unsigned*)(hbuf + 4 * DH);
        hipMemsetAsync(hbuf, 0,
                       4 * DH * sizeof(float) + 2 * SEQ * sizeof(unsigned),
                       stream);
        proj_gemm_cvt<<<gemm_blocks, 256, 0, stream>>>(
            x, fw_ih, fb_ih, fb_hh, u_f, 0);
        proj_gemm_cvt<<<gemm_blocks, 256, 0, stream>>>(
            x, bw_ih, bb_ih, bb_hh, u_b, 1);
    }

    hipFuncSetAttribute((const void*)rnn_scan,
                        hipFuncAttributeMaxDynamicSharedMemorySize,
                        (int)SCAN_LDS);
    rnn_scan<<<dim3(NWG, 2), 256, SCAN_LDS, stream>>>(fw_hh, bw_hh, u_f, out,
                                                      hbuf, cnt);
    (void)in_sizes; (void)n_in; (void)out_size;
}
